// Attention_70781061038144
// MI455X (gfx1250) — matrix operand order repaired and checked by reading, not compiled, not run on hardware
//
#include <hip/hip_runtime.h>
#include <hip/hip_bf16.h>

// ---------------------------------------------------------------------------
// Fused masked MHA for MI455X (gfx1250, wave32, WMMA).
//   B=4, S=2048, F=512, H=8, D=64.
// Kernel 1: QKV projection GEMM (bf16 WMMA, fp32 accum), software-pipelined
//           LDS staging; scatters bf16 Q/K/V into d_ws as [h][b*S+s][d].
// Kernel 2: flash-attention, bf16 WMMA for QK^T and P*V, fp32 online softmax,
//           double-buffered K/mask tiles via GLOBAL_LOAD_ASYNC_TO_LDS_B128
//           (ASYNCcnt). fp32->bf16 via compiler fptrunc (native cvt if avail).
// ---------------------------------------------------------------------------

typedef unsigned short u16;
typedef __bf16  bf16x16 __attribute__((ext_vector_type(16)));
typedef __bf16  bf16x2  __attribute__((ext_vector_type(2)));
typedef float   f32x8   __attribute__((ext_vector_type(8)));
typedef unsigned int u32x4 __attribute__((ext_vector_type(4)));
typedef int     i32x4   __attribute__((ext_vector_type(4)));

#define S_LEN 2048
#define NH    8
#define DH    64
#define NB    4
#define BQ    128   // query rows per attention block
#define BK    32    // keys per flash iteration
#define NIT   (S_LEN / BK)

#if defined(__gfx1250__) && __has_builtin(__builtin_amdgcn_global_load_async_to_lds_b128)
#define USE_ASYNC_LDS 1
#else
#define USE_ASYNC_LDS 0
#endif

#if USE_ASYNC_LDS
// builtin prototype (probe-confirmed by round-2 diagnostic): vV4i*1V4i*3IiIi
typedef __attribute__((address_space(1))) i32x4 gas_i32x4;
typedef __attribute__((address_space(3))) i32x4 las_i32x4;

__device__ __forceinline__ void async_b128(const void* gptr, void* lptr) {
    // per-lane 16B copy, global -> LDS, tracked by ASYNCcnt
    __builtin_amdgcn_global_load_async_to_lds_b128(
        (gas_i32x4*)(unsigned long long)(uintptr_t)gptr,  // int->AS1 ptr (same repr)
        (las_i32x4*)lptr,                                 // generic->AS3 addrspacecast
        0, 0);
}
__device__ __forceinline__ void wait_async0() {
#if __has_builtin(__builtin_amdgcn_s_wait_asynccnt)
    __builtin_amdgcn_s_wait_asynccnt(0);
#else
    asm volatile("s_wait_asynccnt 0x0" ::: "memory");
#endif
}
#endif

// Pack two fp32 -> {lo,hi} bf16 in a u32, via fptrunc (RNE). The backend
// selects the native packed convert when the target has one; otherwise it
// legalizes to the usual round-and-shift sequence.
__device__ __forceinline__ unsigned pk2bf(float a, float b) {
#if __has_builtin(__builtin_amdgcn_cvt_pk_bf16_f32)
    union { bf16x2 v; unsigned u; } x;
    x.v = __builtin_amdgcn_cvt_pk_bf16_f32(a, b);
    return x.u;
#else
    union { bf16x2 v; unsigned u; } x;
    x.v[0] = (__bf16)a;
    x.v[1] = (__bf16)b;
    return x.u;
#endif
}
__device__ __forceinline__ u16 f2bf(float f) {
    union { __bf16 v; u16 u; } x;
    x.v = (__bf16)f;
    return x.u;
}

// Two 16-byte loads -> one 16x32-bf16 fragment (ISA 7.12.2 layouts).
__device__ __forceinline__ bf16x16 load_frag2(const u16* p0, const u16* p1) {
    union { u32x4 u[2]; bf16x16 v; } f;
    f.u[0] = *reinterpret_cast<const u32x4*>(p0);
    f.u[1] = *reinterpret_cast<const u32x4*>(p1);
    return f.v;
}

__device__ __forceinline__ f32x8 wmma_bf16(bf16x16 a, bf16x16 b, f32x8 c) {
    return __builtin_amdgcn_wmma_f32_16x16x32_bf16(
        false, a, false, b, (short)0, c, false, false);
}

__device__ __forceinline__ float rmax16(float x) {
#pragma unroll
    for (int m = 1; m < 16; m <<= 1) x = fmaxf(x, __shfl_xor(x, m, 32));
    return x;
}
__device__ __forceinline__ float rsum16(float x) {
#pragma unroll
    for (int m = 1; m < 16; m <<= 1) x += __shfl_xor(x, m, 32);
    return x;
}

// ---------------------------------------------------------------------------
// Kernel 1: proj[8192,1536] = X[8192,512] @ W[512,1536] + bias, scattered to
// bf16 Q/K/V with column decomposition c = d*24 + which*8 + h.
// Block tile: M=128 (8 waves x 16 rows), N=64. K chunks of 32, reg-pipelined.
// ---------------------------------------------------------------------------
__global__ __launch_bounds__(256) void qkv_proj_kernel(
    const float* __restrict__ X, const float* __restrict__ W,
    const float* __restrict__ bias, u16* __restrict__ qkv)
{
    __shared__ __align__(16) u16 smX[128 * 32];   // bf16 X tile, row-major
    __shared__ __align__(16) u16 smWt[64 * 32];   // bf16 W tile, transposed [n][k]

    const int tid = threadIdx.x;
    const int lane = tid & 31, wv = tid >> 5;
    const int n16 = lane & 15, lhi = lane >> 4;

    const int mt = blockIdx.x & 63;          // 64 M tiles
    const int nt = blockIdx.x >> 6;          // 24 N tiles
    const int m0 = mt * 128, n0 = nt * 64;

    // staging-thread mappings
    const int xr = tid >> 1, xcg = (tid & 1) * 16;       // X: 16 floats/thread
    const int wk = tid >> 3, wng = (tid & 7) * 8;        // W: 8 floats/thread
    const float* Xrow = X + (size_t)(m0 + xr) * 512 + xcg;
    const float* Wrow = W + n0 + wng + (size_t)wk * 1536;

    f32x8 acc[4];
#pragma unroll
    for (int nc = 0; nc < 4; ++nc)
#pragma unroll
        for (int i = 0; i < 8; ++i) acc[nc][i] = 0.f;

    float xg[16], wg[8];
    auto issue = [&](int kc) {
#pragma unroll
        for (int i = 0; i < 16; ++i) xg[i] = Xrow[kc + i];
#pragma unroll
        for (int i = 0; i < 8; ++i) wg[i] = Wrow[(size_t)kc * 1536 + i];
    };
    auto commit = [&]() {
        u32x4 pk[2];
#pragma unroll
        for (int q = 0; q < 2; ++q)
#pragma unroll
            for (int i = 0; i < 4; ++i)
                pk[q][i] = pk2bf(xg[q * 8 + 2 * i], xg[q * 8 + 2 * i + 1]);
        u32x4* dst = reinterpret_cast<u32x4*>(&smX[xr * 32 + xcg]);
        dst[0] = pk[0]; dst[1] = pk[1];
#pragma unroll
        for (int i = 0; i < 8; i += 2) {
            unsigned p = pk2bf(wg[i], wg[i + 1]);      // rows ng+i, ng+i+1
            smWt[(wng + i) * 32 + wk]     = (u16)p;
            smWt[(wng + i + 1) * 32 + wk] = (u16)(p >> 16);
        }
    };

    issue(0);
    for (int kc = 0; kc < 512; kc += 32) {
        commit();
        __syncthreads();
        if (kc + 32 < 512) issue(kc + 32);   // overlap next global loads w/ WMMAs

        // A fragment (16x32 bf16): k = lhi*16 + e (contiguous)
        const u16* ar = &smX[(wv * 16 + n16) * 32];
        bf16x16 af = load_frag2(ar + lhi * 16, ar + lhi * 16 + 8);
#pragma unroll
        for (int nc = 0; nc < 4; ++nc) {
            // B fragment (32x16 bf16): lane=N, k = lhi*16 + e (contiguous)
            const u16* br = &smWt[(nc * 16 + n16) * 32 + lhi * 16];
            acc[nc] = wmma_bf16(af, load_frag2(br, br + 8), acc[nc]);
        }
        __syncthreads();
    }

    // Epilogue: bias + scatter bf16 into Q/K/V [h][m][d], m = b*S+s.
    const size_t HMD = (size_t)NH * (NB * S_LEN) * DH;   // 4,194,304
#pragma unroll
    for (int nc = 0; nc < 4; ++nc) {
        int c = n0 + nc * 16 + n16;
        int d = c / 24, rr = c - d * 24;
        int which = rr >> 3, h = rr & 7;
        float bv = bias[c];
        u16* dst = qkv + (size_t)which * HMD
                 + (size_t)h * (size_t)(NB * S_LEN) * DH + d;
#pragma unroll
        for (int r = 0; r < 8; r += 2) {
            int m = m0 + wv * 16 + r + 8 * lhi;          // C/D layout rows m, m+1
            unsigned p = pk2bf(acc[nc][r] + bv, acc[nc][r + 1] + bv);
            dst[(size_t)m * DH]       = (u16)p;
            dst[(size_t)(m + 1) * DH] = (u16)(p >> 16);
        }
    }
}

// ---------------------------------------------------------------------------
// Kernel 2: flash attention. Block = 128 query rows of one (b,h); 8 waves.
// Double-buffered tiles; K & mask go global->LDS via async copies (ASYNCcnt),
// V is register-staged (needs a transpose on the LDS store).
// ---------------------------------------------------------------------------
__global__ __launch_bounds__(256) void flash_attn_kernel(
    const u16* __restrict__ qkv, const unsigned char* __restrict__ Amask,
    float* __restrict__ out)
{
    __shared__ __align__(16) u16 smK[2][BK * DH];            // K tile [key][d]
    __shared__ __align__(16) u16 smVt[2][DH * BK];           // V tile transposed
    __shared__ __align__(16) unsigned char smM[2][BQ * BK];  // mask tile
    __shared__ __align__(16) u16 smP[8][16 * BK];            // per-wave P bounce

    const int tid = threadIdx.x;
    const int lane = tid & 31, wv = tid >> 5;
    const int n16 = lane & 15, lhi = lane >> 4;

    const int rowblk = blockIdx.x & 15;                  // 16 row blocks / (b,h)
    const int bh = blockIdx.x >> 4;
    const int b = bh >> 3, h = bh & 7;
    const int q0 = rowblk * BQ;

    const size_t HMD = (size_t)NH * (NB * S_LEN) * DH;
    const u16* Qb = qkv;
    const u16* Kb = qkv + HMD;
    const u16* Vb = qkv + 2 * HMD;
    const size_t base_hb = ((size_t)h * NB + b) * (size_t)S_LEN * DH;

    // Q fragments for this wave's 16 rows (resident whole kernel).
    bf16x16 qf[2];
    {
        const u16* qrow = Qb + base_hb + (size_t)(q0 + wv * 16 + n16) * DH;
#pragma unroll
        for (int c = 0; c < 2; ++c)
            qf[c] = load_frag2(qrow + c * 32 + lhi * 16,
                               qrow + c * 32 + lhi * 16 + 8);
    }

    float mrow[8], lrow[8];
    f32x8 o[4];
#pragma unroll
    for (int r = 0; r < 8; ++r) { mrow[r] = -3.0e38f; lrow[r] = 0.f; }
#pragma unroll
    for (int nc = 0; nc < 4; ++nc)
#pragma unroll
        for (int i = 0; i < 8; ++i) o[nc][i] = 0.f;

    const float scale = 0.35355339059327373f;            // 1/sqrt(H=8)

    // staging-thread mappings
    const int r32 = tid >> 3, c8 = (tid & 7) << 3;       // K/V: 8 halves/thread
    const int qr = tid >> 1, kk = (tid & 1) << 4;        // mask: 16 B/thread
    const u16* Ksrc = Kb + base_hb + (size_t)r32 * DH + c8;
    const u16* Vsrc = Vb + base_hb + (size_t)r32 * DH + c8;
    const unsigned char* Msrc = Amask + (size_t)b * S_LEN * S_LEN
                              + (size_t)(q0 + qr) * S_LEN + kk;

    u32x4 stV;
#if !USE_ASYNC_LDS
    u32x4 stK, stM;
#endif
    auto issue = [&](int j0, int buf) {
#if USE_ASYNC_LDS
        async_b128(Ksrc + (size_t)j0 * DH, &smK[buf][r32 * DH + c8]);
        async_b128(Msrc + j0,              &smM[buf][qr * BK + kk]);
#else
        stK = *reinterpret_cast<const u32x4*>(Ksrc + (size_t)j0 * DH);
        stM = *reinterpret_cast<const u32x4*>(Msrc + j0);
#endif
        stV = *reinterpret_cast<const u32x4*>(Vsrc + (size_t)j0 * DH);
    };
    auto commit = [&](int buf) {
#if USE_ASYNC_LDS
        wait_async0();                              // my async writes landed
#else
        *reinterpret_cast<u32x4*>(&smK[buf][r32 * DH + c8]) = stK;
        *reinterpret_cast<u32x4*>(&smM[buf][qr * BK + kk])  = stM;
#endif
        const u16* vh = reinterpret_cast<const u16*>(&stV);
#pragma unroll
        for (int i = 0; i < 8; ++i) smVt[buf][(c8 + i) * BK + r32] = vh[i];
    };

    issue(0, 0);
    for (int j = 0; j < NIT; ++j) {
        const int buf = j & 1;
        commit(buf);
        __syncthreads();                            // tiles visible to all waves
        if (j + 1 < NIT) issue((j + 1) * BK, buf ^ 1);   // overlap with compute

        // scores: S[16 x 32] = Q(16x64) * K^T(64x32), split into 2 N-halves
        f32x8 s0, s1;
#pragma unroll
        for (int i = 0; i < 8; ++i) { s0[i] = 0.f; s1[i] = 0.f; }
#pragma unroll
        for (int c = 0; c < 2; ++c) {
            const u16* b0 = &smK[buf][(size_t)n16 * DH + c * 32 + lhi * 16];
            const u16* b1 = &smK[buf][(size_t)(16 + n16) * DH + c * 32 + lhi * 16];
            s0 = wmma_bf16(qf[c], load_frag2(b0, b0 + 8), s0);
            s1 = wmma_bf16(qf[c], load_frag2(b1, b1 + 8), s1);
        }

        // mask + online softmax (rows map to (r, lhi); N = lanes 0..15)
        float pr0[8], pr1[8], alpha[8];
#pragma unroll
        for (int r = 0; r < 8; ++r) {
            int ql = wv * 16 + r + 8 * lhi;
            float v0 = s0[r] * scale, v1 = s1[r] * scale;
            if (!smM[buf][ql * BK + n16])      v0 = -1.0e9f;  // reference -INF
            if (!smM[buf][ql * BK + 16 + n16]) v1 = -1.0e9f;
            float rm = rmax16(fmaxf(v0, v1));
            float mn = fmaxf(mrow[r], rm);
            float a = __expf(mrow[r] - mn);
            pr0[r] = __expf(v0 - mn);
            pr1[r] = __expf(v1 - mn);
            lrow[r] = lrow[r] * a + rsum16(pr0[r] + pr1[r]);
            mrow[r] = mn; alpha[r] = a;
        }
#pragma unroll
        for (int nc = 0; nc < 4; ++nc)
#pragma unroll
            for (int r = 0; r < 8; ++r) o[nc][r] *= alpha[r];

        // P: C/D layout -> A-fragment layout via per-wave LDS (in-order DS ops)
        // pk-convert row pairs (r, r+1) -> LDS rows m, m+1.
        u16* pw = &smP[wv][0];
#pragma unroll
        for (int r = 0; r < 8; r += 2) {
            int m = r + 8 * lhi;
            unsigned p0 = pk2bf(pr0[r], pr0[r + 1]);
            unsigned p1 = pk2bf(pr1[r], pr1[r + 1]);
            pw[m * BK + n16]            = (u16)p0;
            pw[(m + 1) * BK + n16]      = (u16)(p0 >> 16);
            pw[m * BK + 16 + n16]       = (u16)p1;
            pw[(m + 1) * BK + 16 + n16] = (u16)(p1 >> 16);
        }
        bf16x16 pf = load_frag2(&pw[n16 * BK + lhi * 16],
                                &pw[n16 * BK + lhi * 16 + 8]);

        // O[16x64] += P(16x32) * V(32x64)
#pragma unroll
        for (int nc = 0; nc < 4; ++nc) {
            const u16* vr = &smVt[buf][(size_t)(nc * 16 + n16) * BK + lhi * 16];
            o[nc] = wmma_bf16(pf, load_frag2(vr, vr + 8), o[nc]);
        }
        __syncthreads();                            // buf free for j+2's async
    }

    // Y[b,s,d*H+h] = O / l
#pragma unroll
    for (int nc = 0; nc < 4; ++nc) {
#pragma unroll
        for (int r = 0; r < 8; ++r) {
            int s = q0 + wv * 16 + r + 8 * lhi;
            int d = nc * 16 + n16;
            out[((size_t)b * S_LEN + s) * (NH * DH) + d * NH + h] = o[nc][r] / lrow[r];
        }
    }
}

// ---------------------------------------------------------------------------
extern "C" void kernel_launch(void* const* d_in, const int* in_sizes, int n_in,
                              void* d_out, int out_size, void* d_ws, size_t ws_size,
                              hipStream_t stream)
{
    const float*         X    = (const float*)d_in[0];          // [4,2048,512]
    const unsigned char* A    = (const unsigned char*)d_in[1];  // bool [4,2048,2048]
    const float*         W    = (const float*)d_in[2];          // [512,1536]
    const float*         bias = (const float*)d_in[3];          // [1536]
    float*               out  = (float*)d_out;                  // [4,2048,512]

    u16* qkv = (u16*)d_ws;   // 3 * 8*8192*64 bf16 = 24 MB scratch

    qkv_proj_kernel<<<64 * 24, 256, 0, stream>>>(X, W, bias, qkv);
    flash_attn_kernel<<<32 * 16, 256, 0, stream>>>(qkv, A, out);
}
